// QueryBasedDecoder_19069654794261
// MI455X (gfx1250) — matrix operand order repaired and checked by reading, not compiled, not run on hardware
//
#include <hip/hip_runtime.h>
#include <math.h>

// ---------------------------------------------------------------------------
// Query-based decoder (DETR-style, 1-D temporal) for MI455X / gfx1250.
//
// All dense projections use V_WMMA_F32_16X16X32_BF16. Activations are
// converted to bf16 once by their producer kernels, so GEMM inner loops are
// pure b128-load + v_wmma. Each GEMM wave computes a 32x64 C tile
// (8 accumulators). The k-loop is software-pipelined with TWO named fragment
// sets (ping-pong, unrolled x2) so latency is hidden with zero register
// rotation. The dominant pool GEMM (K = NS*R*D = 16384) fuses ROI-align lerp
// into the A-fragment build so the 125 MB feature pyramid stays L2-resident
// (192 MB L2) and the 630 MB "pooled" tensor is never materialized.
// ---------------------------------------------------------------------------

namespace {

constexpr int Dm   = 256;          // model dim
constexpr int NHh  = 8;            // heads
constexpr int DH   = 32;           // head dim
constexpr int FFd  = 1024;         // FFN dim
constexpr int NQq  = 300;          // queries
constexpr int NLl  = 4;            // layers
constexpr int Rr   = 16;           // ROI samples
constexpr int NSs  = 4;            // scales
constexpr int Bb   = 32;           // batch
constexpr int MR   = Bb * NQq;     // 9600 activation rows
constexpr int KP   = NSs * Rr * Dm;// 16384 pool-GEMM K
constexpr int TMEM = 256;          // feat3 (memory) length

typedef __attribute__((ext_vector_type(16))) __bf16         v16bf;
typedef __attribute__((ext_vector_type(16))) unsigned short v16u;
typedef __attribute__((ext_vector_type(8)))  unsigned short v8u;
typedef __attribute__((ext_vector_type(8)))  float          v8f;

__device__ __forceinline__ unsigned short f2bf(float f) {
  unsigned int u = __float_as_uint(f);
  u += 0x7FFFu + ((u >> 16) & 1u);            // round-to-nearest-even
  return (unsigned short)(u >> 16);
}

__device__ __forceinline__ v16bf cat16(v8u lo, v8u hi) {
  return __builtin_bit_cast(
      v16bf, __builtin_shufflevector(lo, hi, 0, 1, 2, 3, 4, 5, 6, 7, 8, 9, 10,
                                     11, 12, 13, 14, 15));
}

// A-fragment load: two contiguous 16-byte runs (ISA 16-bit A layout).
__device__ __forceinline__ v16bf load_a(const unsigned short* p, int off) {
  return cat16(*(const v8u*)(p + off), *(const v8u*)(p + off + 16));
}
// B-fragment load: two contiguous 16-byte runs (same k order as A).
__device__ __forceinline__ v16bf load_b(const unsigned short* p, size_t off) {
  return cat16(*(const v8u*)(p + off), *(const v8u*)(p + off + 16));
}

__device__ __forceinline__ v8f wmma_bf16(v16bf a, v16bf b, v8f c) {
  return __builtin_amdgcn_wmma_f32_16x16x32_bf16(false, a, false, b, (short)0,
                                                 c, false, false);
}

__device__ __forceinline__ float wave_sum(float v) {
#pragma unroll
  for (int m = 16; m >= 1; m >>= 1) v += __shfl_xor(v, m, 32);
  return v;
}

// --------------------------- f32 -> bf16 conversion ------------------------
__global__ void cvt_bf16_kernel(const float* __restrict__ s,
                                unsigned short* __restrict__ d, int n) {
  int i = blockIdx.x * 256 + threadIdx.x;
  if (i < n) d[i] = f2bf(s[i]);
}

// --------------------------- generic bf16 WMMA GEMM -------------------------
// C[M,N] = act(A[M,K] (bf16) @ W[N,K]^T (bf16) + bias).
// Wave tile: 32(M) x 64(N) = 8 v8f accumulators. K-loop is ping-pong
// pipelined over 64-k chunks (K is always a multiple of 64 here):
//   load set-B(k+32); WMMA set-A(k); load set-A(k+64); WMMA set-B(k+32)
// Fragment layouts per ISA tables:
//   A: lane m=l%16, elems e -> K = kb + e + 8*(e>>3) + 8*(l>>4)
//   B: lane n=l%16, elems e -> K = kb + e + 8*(e>>3) + 8*(l>>4)
template <int RELU, int OUTBF>
__global__ __launch_bounds__(256) void gemm_bf16_kernel(
    const unsigned short* __restrict__ A, const unsigned short* __restrict__ W,
    const float* __restrict__ bias, float* __restrict__ C,
    unsigned short* __restrict__ Cb, int M, int N, int K) {
  const int lane = threadIdx.x & 31;
  const int wid  = blockIdx.x * 8 + (threadIdx.x >> 5);
  const int stN  = N >> 6;
  if (wid >= (M >> 5) * stN) return;
  const int tm = wid / stN;
  const int tn = wid - tm * stN;
  const int lo = lane & 15;
  const int hi = lane >> 4;
  const unsigned short* ap0 = A + (size_t)(tm * 32 + lo) * K + hi * 8;
  const unsigned short* ap1 = ap0 + (size_t)16 * K;
  const unsigned short* wpt = W + (size_t)(tn * 64 + lo) * K + hi * 8;
  const size_t wstride = (size_t)16 * K;   // between 16-col B fragments
  v8f acc[2][4] = {};
  // prologue: set-A fragments for kb = 0
  v16bf a0A = load_a(ap0, 0);
  v16bf a1A = load_a(ap1, 0);
  v16bf bA[4];
#pragma unroll
  for (int nt = 0; nt < 4; ++nt) bA[nt] = load_b(wpt, wstride * nt);
  v16bf a0B, a1B, bB[4];
  int kb = 0;
  for (; kb + 64 < K; kb += 64) {
    // stage set-B (kb+32); WMMAs below cover the latency
    a0B = load_a(ap0, kb + 32);
    a1B = load_a(ap1, kb + 32);
#pragma unroll
    for (int nt = 0; nt < 4; ++nt) bB[nt] = load_b(wpt, wstride * nt + kb + 32);
    __builtin_prefetch(ap0 + kb + 96, 0, 1);  // global_prefetch_b8
    __builtin_prefetch(wpt + kb + 96, 0, 1);
#pragma unroll
    for (int nt = 0; nt < 4; ++nt) {
      acc[0][nt] = wmma_bf16(a0A, bA[nt], acc[0][nt]);
      acc[1][nt] = wmma_bf16(a1A, bA[nt], acc[1][nt]);
    }
    // stage set-A (kb+64)
    a0A = load_a(ap0, kb + 64);
    a1A = load_a(ap1, kb + 64);
#pragma unroll
    for (int nt = 0; nt < 4; ++nt) bA[nt] = load_b(wpt, wstride * nt + kb + 64);
#pragma unroll
    for (int nt = 0; nt < 4; ++nt) {
      acc[0][nt] = wmma_bf16(a0B, bB[nt], acc[0][nt]);
      acc[1][nt] = wmma_bf16(a1B, bB[nt], acc[1][nt]);
    }
  }
  // tail chunk (kb + 64 == K): set-A holds kb, stage set-B at kb+32
  a0B = load_a(ap0, kb + 32);
  a1B = load_a(ap1, kb + 32);
#pragma unroll
  for (int nt = 0; nt < 4; ++nt) bB[nt] = load_b(wpt, wstride * nt + kb + 32);
#pragma unroll
  for (int nt = 0; nt < 4; ++nt) {
    acc[0][nt] = wmma_bf16(a0A, bA[nt], acc[0][nt]);
    acc[1][nt] = wmma_bf16(a1A, bA[nt], acc[1][nt]);
  }
#pragma unroll
  for (int nt = 0; nt < 4; ++nt) {
    acc[0][nt] = wmma_bf16(a0B, bB[nt], acc[0][nt]);
    acc[1][nt] = wmma_bf16(a1B, bB[nt], acc[1][nt]);
  }
#pragma unroll
  for (int mt = 0; mt < 2; ++mt) {
#pragma unroll
    for (int nt = 0; nt < 4; ++nt) {
      const int ncol = tn * 64 + nt * 16 + lo;
      const float bn = bias ? bias[ncol] : 0.0f;
#pragma unroll
      for (int r = 0; r < 8; ++r) {
        const int m = tm * 32 + mt * 16 + r + 8 * hi;
        float v = acc[mt][nt][r] + bn;
        if (RELU) v = fmaxf(v, 0.0f);
        if (OUTBF) Cb[(size_t)m * N + ncol] = f2bf(v);
        else       C[(size_t)m * N + ncol] = v;
      }
    }
  }
}

// --------------- fused ROI-align + pool GEMM (layer 0 of pool MLP) ----------
// Cb[MR,256](bf16) = relu( A_roi[MR,16384] @ W0[256,16384]^T + b0 ), where
// A_roi[(b,q), s*4096+r*256+d] = lerp(feat_s[b,i0,d], feat_s[b,i1,d], w).
// Wave tile 16(M) x 64(N). Weight fragments are ping-pong pipelined over
// 64-k chunks; the fused ROI-align A-fragment build (f32 lerp + cvt VALU)
// overlaps the staged weight loads.
__global__ __launch_bounds__(256) void pool_gemm_kernel(
    const float* __restrict__ f0, const float* __restrict__ f1,
    const float* __restrict__ f2, const float* __restrict__ f3,
    const int* __restrict__ ri0, const int* __restrict__ ri1,
    const float* __restrict__ rw, const unsigned short* __restrict__ W,
    const float* __restrict__ bias, unsigned short* __restrict__ Cb) {
  const int lane = threadIdx.x & 31;
  const int wid  = blockIdx.x * 8 + (threadIdx.x >> 5);
  const int stN  = Dm >> 6;                       // 4
  if (wid >= (MR >> 4) * stN) return;
  const int tm = wid / stN;
  const int tn = wid - tm * stN;
  const int lo = lane & 15;
  const int hi = lane >> 4;
  const int mrow = tm * 16 + lo;
  const int bq_b = mrow / NQq;
  const int bq_q = mrow - bq_b * NQq;
  const unsigned short* wpt = W + (size_t)(tn * 64 + lo) * KP + hi * 8;
  const size_t wstride = (size_t)16 * KP;

  // fused ROI-align A-fragment build for a 32-k tile starting at kb
  auto build_a = [&](int kb) -> v16bf {
    const int s     = kb >> 12;
    const int r     = (kb >> 8) & 15;
    const int dbase = kb & 255;
    const int ridx  = ((bq_b * NQq + bq_q) * NSs + s) * Rr + r;
    const int i0 = ri0[ridx];
    const int i1 = ri1[ridx];
    const float w = rw[ridx];
    const float* fp; int T;
    switch (s) {
      case 0:  fp = f0; T = 2048; break;
      case 1:  fp = f1; T = 1024; break;
      case 2:  fp = f2; T = 512;  break;
      default: fp = f3; T = 256;  break;
    }
    const float* p0 = fp + ((size_t)(bq_b * T + i0)) * Dm + dbase + hi * 8;
    const float* p1 = fp + ((size_t)(bq_b * T + i1)) * Dm + dbase + hi * 8;
    v16u au;
#pragma unroll
    for (int half = 0; half < 2; ++half) {
      const int off = half * 16;
      float4 x0 = *(const float4*)(p0 + off);
      float4 x1 = *(const float4*)(p0 + off + 4);
      float4 y0 = *(const float4*)(p1 + off);
      float4 y1 = *(const float4*)(p1 + off + 4);
      au[half * 8 + 0] = f2bf(x0.x + w * (y0.x - x0.x));
      au[half * 8 + 1] = f2bf(x0.y + w * (y0.y - x0.y));
      au[half * 8 + 2] = f2bf(x0.z + w * (y0.z - x0.z));
      au[half * 8 + 3] = f2bf(x0.w + w * (y0.w - x0.w));
      au[half * 8 + 4] = f2bf(x1.x + w * (y1.x - x1.x));
      au[half * 8 + 5] = f2bf(x1.y + w * (y1.y - x1.y));
      au[half * 8 + 6] = f2bf(x1.z + w * (y1.z - x1.z));
      au[half * 8 + 7] = f2bf(x1.w + w * (y1.w - x1.w));
    }
    return __builtin_bit_cast(v16bf, au);
  };

  v8f acc[4] = {};
  v16bf bA[4];
#pragma unroll
  for (int nt = 0; nt < 4; ++nt) bA[nt] = load_b(wpt, wstride * nt);
  v16bf bB[4];
  int kb = 0;
  for (; kb + 64 < KP; kb += 64) {
    // stage set-B weights (kb+32); A build + WMMAs cover the latency
#pragma unroll
    for (int nt = 0; nt < 4; ++nt) bB[nt] = load_b(wpt, wstride * nt + kb + 32);
    __builtin_prefetch(wpt + kb + 96, 0, 1);      // global_prefetch_b8
    {
      const v16bf af = build_a(kb);
#pragma unroll
      for (int nt = 0; nt < 4; ++nt) acc[nt] = wmma_bf16(af, bA[nt], acc[nt]);
    }
    // stage set-A weights (kb+64)
#pragma unroll
    for (int nt = 0; nt < 4; ++nt) bA[nt] = load_b(wpt, wstride * nt + kb + 64);
    {
      const v16bf af = build_a(kb + 32);
#pragma unroll
      for (int nt = 0; nt < 4; ++nt) acc[nt] = wmma_bf16(af, bB[nt], acc[nt]);
    }
  }
  // tail chunk (kb + 64 == KP)
#pragma unroll
  for (int nt = 0; nt < 4; ++nt) bB[nt] = load_b(wpt, wstride * nt + kb + 32);
  {
    const v16bf af = build_a(kb);
#pragma unroll
    for (int nt = 0; nt < 4; ++nt) acc[nt] = wmma_bf16(af, bA[nt], acc[nt]);
  }
  {
    const v16bf af = build_a(kb + 32);
#pragma unroll
    for (int nt = 0; nt < 4; ++nt) acc[nt] = wmma_bf16(af, bB[nt], acc[nt]);
  }
#pragma unroll
  for (int nt = 0; nt < 4; ++nt) {
    const int ncol = tn * 64 + nt * 16 + lo;
    const float bn = bias[ncol];
#pragma unroll
    for (int rI = 0; rI < 8; ++rI) {
      const int m = tm * 16 + rI + 8 * hi;
      Cb[(size_t)m * Dm + ncol] = f2bf(fmaxf(acc[nt][rI] + bn, 0.0f));
    }
  }
}

// --------------------------- flash-style attention -------------------------
// One (b,h) per blockIdx.x, one query per thread (grid.y covers NQ).
// K/V chunks of 128 rows in 32 KB LDS; online softmax per thread.
// Output is written bf16-only (consumed only by the out-projection GEMM).
constexpr int ACHUNK = 128;
__global__ __launch_bounds__(128) void attn_kernel(
    const float* __restrict__ Q, int ldq, const float* __restrict__ Kp, int ldk,
    const float* __restrict__ Vp, int ldv, unsigned short* __restrict__ Ob,
    int ldo, int Tq, int Tk, float scale) {
  __shared__ float sK[ACHUNK * DH];
  __shared__ float sV[ACHUNK * DH];
  const int b = blockIdx.x / NHh;
  const int h = blockIdx.x % NHh;
  const int q = blockIdx.y * 128 + threadIdx.x;
  const bool act = (q < Tq);
  float qv[DH], accv[DH];
  float mx = -1e30f, ssum = 0.0f;
  if (act) {
    const float* qp = Q + ((size_t)(b * Tq + q)) * ldq + h * DH;
#pragma unroll
    for (int d = 0; d < DH; ++d) { qv[d] = qp[d]; accv[d] = 0.0f; }
  }
  for (int j0 = 0; j0 < Tk; j0 += ACHUNK) {
    const int cn = min(ACHUNK, Tk - j0);
    __syncthreads();
    for (int idx = threadIdx.x; idx < cn * DH; idx += 128) {
      const int row = idx / DH, c = idx % DH;
      sK[idx] = Kp[((size_t)(b * Tk + j0 + row)) * ldk + h * DH + c];
      sV[idx] = Vp[((size_t)(b * Tk + j0 + row)) * ldv + h * DH + c];
    }
    __syncthreads();
    if (act) {
      for (int j = 0; j < cn; ++j) {
        float s = 0.0f;
#pragma unroll
        for (int d = 0; d < DH; ++d) s += qv[d] * sK[j * DH + d];
        s *= scale;
        const float nm   = fmaxf(mx, s);
        const float fold = __expf(mx - nm);
        const float p    = __expf(s - nm);
        ssum = ssum * fold + p;
#pragma unroll
        for (int d = 0; d < DH; ++d)
          accv[d] = accv[d] * fold + p * sV[j * DH + d];
        mx = nm;
      }
    }
  }
  if (act) {
    const float inv = 1.0f / ssum;
    unsigned short* op = Ob + ((size_t)(b * Tq + q)) * ldo + h * DH;
#pragma unroll
    for (int d = 0; d < DH; ++d) op[d] = f2bf(accv[d] * inv);
  }
}

// ------------------- residual add + LayerNorm (f32 + bf16 mirror) ----------
__global__ __launch_bounds__(256) void add_ln_kernel(
    const float* __restrict__ X, const float* __restrict__ P,
    const float* __restrict__ g, const float* __restrict__ b,
    float* __restrict__ O, unsigned short* __restrict__ Ob, int rows) {
  const int lane = threadIdx.x & 31;
  const int row  = blockIdx.x * 8 + (threadIdx.x >> 5);
  if (row >= rows) return;
  const float* xr = X + (size_t)row * Dm;
  const float* pr = P + (size_t)row * Dm;
  float v[8];
  float s = 0.0f;
#pragma unroll
  for (int i = 0; i < 8; ++i) {
    const int c = lane + 32 * i;
    v[i] = xr[c] + pr[c];
    s += v[i];
  }
  s = wave_sum(s);
  const float mean = s * (1.0f / Dm);
  float var = 0.0f;
#pragma unroll
  for (int i = 0; i < 8; ++i) { const float d = v[i] - mean; var += d * d; }
  var = wave_sum(var) * (1.0f / Dm);
  const float rs = rsqrtf(var + 1e-5f);
  float* orow = O + (size_t)row * Dm;
  unsigned short* obrow = Ob + (size_t)row * Dm;
#pragma unroll
  for (int i = 0; i < 8; ++i) {
    const int c = lane + 32 * i;
    const float y = (v[i] - mean) * rs * g[c] + b[c];
    orow[c] = y;
    obrow[c] = f2bf(y);
  }
}

// ---------- small head: out[row, j] = dot(X[row,:], W[j,:]) + bias[j] -------
__global__ __launch_bounds__(256) void head_kernel(
    const float* __restrict__ X, const float* __restrict__ W,
    const float* __restrict__ bias, float* __restrict__ Out,
    int rows, int ncols) {
  const int lane = threadIdx.x & 31;
  const int row  = blockIdx.x * 8 + (threadIdx.x >> 5);
  if (row >= rows) return;
  const float* xr = X + (size_t)row * Dm;
  for (int j = 0; j < ncols; ++j) {
    const float* wr = W + j * Dm;
    float s = 0.0f;
#pragma unroll
    for (int i = 0; i < 8; ++i) { const int c = lane + 32 * i; s += xr[c] * wr[c]; }
    s = wave_sum(s);
    if (lane == 0) Out[(size_t)row * ncols + j] = s + bias[j];
  }
}

// ------------- reference update, proposals, ROI interpolation tables --------
__device__ __forceinline__ float inv_sig(float x, float eps) {
  x = fminf(fmaxf(x, 0.0f), 1.0f);
  return logf(fmaxf(x, eps) / fmaxf(1.0f - x, eps));
}
__global__ void ref_update_kernel(
    const float* __restrict__ offs, const float* __restrict__ refc,
    float* __restrict__ refb, float* __restrict__ prop,
    int* __restrict__ ri0, int* __restrict__ ri1, float* __restrict__ rw,
    int first) {
  const int i = blockIdx.x * 256 + threadIdx.x;
  if (i >= MR) return;
  float o0 = offs[i * 2 + 0], o1 = offs[i * 2 + 1];
  if (first) {
    o0 += inv_sig(refc[i], 1e-12f);
  } else {
    o0 += inv_sig(refb[i * 2 + 0], 1e-5f);
    o1 += inv_sig(refb[i * 2 + 1], 1e-5f);
  }
  const float c = 1.0f / (1.0f + __expf(-o0));
  const float w = 1.0f / (1.0f + __expf(-o1));
  refb[i * 2 + 0] = c;
  refb[i * 2 + 1] = w;
  const float s = c - 0.5f * w, e = c + 0.5f * w;
  prop[i * 2 + 0] = s;
  prop[i * 2 + 1] = e;
  const int Ts[4] = {2048, 1024, 512, 256};
  for (int sc = 0; sc < NSs; ++sc) {
    const float Tf = (float)Ts[sc];
    for (int r = 0; r < Rr; ++r) {
      const float t   = s + (e - s) * ((r + 0.5f) * (1.0f / Rr));
      const float pos = fminf(fmaxf(t * Tf - 0.5f, 0.0f), Tf - 1.0f);
      const int a0 = (int)floorf(pos);
      const int a1 = min(a0 + 1, Ts[sc] - 1);
      const int idx = (i * NSs + sc) * Rr + r;
      ri0[idx] = a0;
      ri1[idx] = a1;
      rw[idx]  = pos - (float)a0;
    }
  }
}

// ----------------- tgt init / query-emb add (f32 + bf16 mirror) -------------
__global__ void init_tgt_kernel(const float* __restrict__ qe,
                                float* __restrict__ tgt,
                                unsigned short* __restrict__ tgtb) {
  const int i = blockIdx.x * 256 + threadIdx.x;
  if (i >= MR * Dm) return;
  const int q = (i / Dm) % NQq;
  const float v = qe[q * Dm + (i & (Dm - 1))];
  tgt[i] = v;
  tgtb[i] = f2bf(v);
}
__global__ void add_qe_kernel(const float* __restrict__ G,
                              const float* __restrict__ qe,
                              float* __restrict__ tgt,
                              unsigned short* __restrict__ tgtb) {
  const int i = blockIdx.x * 256 + threadIdx.x;
  if (i >= MR * Dm) return;
  const int q = (i / Dm) % NQq;
  const float v = G[i] + qe[q * Dm + (i & (Dm - 1))];
  tgt[i] = v;
  tgtb[i] = f2bf(v);
}

// per-layer leaf offsets (JAX pytree, dict keys sorted):
// 0 bbox0.b 1 bbox0.w 2 bbox1.b 3 bbox1.w 4 bbox2.b 5 bbox2.w
// 6 ca.in_b 7 ca.in_w 8 ca.out_b 9 ca.out_w 10 ffn1.b 11 ffn1.w
// 12 ffn2.b 13 ffn2.w 14 ln1.b 15 ln1.g 16 ln2.b 17 ln2.g 18 ln3.b 19 ln3.g
// 20 pool0.b 21 pool0.w 22 pool1.b 23 pool1.w 24 sa.in_b 25 sa.in_w
// 26 sa.out_b 27 sa.out_w 28 score0.b 29 score0.w 30 score1.b 31 score1.w
struct LW { int leaf; int elems; };
static const LW kW[11] = {
    {25, 768 * 256},   // 0 sa.in_w
    {27, 256 * 256},   // 1 sa.out_w
    {7,  768 * 256},   // 2 ca.in_w
    {9,  256 * 256},   // 3 ca.out_w
    {11, 1024 * 256},  // 4 ffn1.w
    {13, 256 * 1024},  // 5 ffn2.w
    {29, 256 * 256},   // 6 score0.w
    {1,  256 * 256},   // 7 bbox0.w
    {3,  256 * 256},   // 8 bbox1.w
    {21, 256 * 16384}, // 9 pool0.w
    {23, 256 * 256},   // 10 pool1.w
};

} // anonymous namespace

extern "C" void kernel_launch(void* const* d_in, const int* in_sizes, int n_in,
                              void* d_out, int out_size, void* d_ws, size_t ws_size,
                              hipStream_t stream) {
  (void)n_in; (void)out_size; (void)ws_size;
  const float* f0   = (const float*)d_in[0];
  const float* f1   = (const float*)d_in[1];
  const float* f2   = (const float*)d_in[2];
  const float* f3   = (const float*)d_in[3];
  const float* refc = (const float*)d_in[4];

  // Param pytree order hedge: sorted keys put 'layers' before 'query_emb',
  // but detect the opposite via the leaf size.
  int pb;
  const float* qe;
  if (in_sizes[5] == NQq * Dm) { qe = (const float*)d_in[5];            pb = 6; }
  else                         { qe = (const float*)d_in[5 + NLl * 32]; pb = 5; }
  auto LP = [&](int li, int leaf) -> const float* {
    return (const float*)d_in[pb + li * 32 + leaf];
  };

  // workspace carve-out
  char* wp = (char*)d_ws;
  size_t cur = 0;
  auto wsalloc = [&](size_t bytes) -> void* {
    void* p = wp + cur;
    cur += (bytes + 255) & ~(size_t)255;
    return p;
  };

  unsigned short* wbf[NLl][11];
  for (int li = 0; li < NLl; ++li)
    for (int s = 0; s < 11; ++s)
      wbf[li][s] = (unsigned short*)wsalloc((size_t)kW[s].elems * 2);

  // f32 activations
  float* X    = (float*)wsalloc((size_t)MR * Dm * 4);         // tgt
  float* QKV  = (float*)wsalloc((size_t)MR * 3 * Dm * 4);     // also HS/H2 later
  float* P    = (float*)wsalloc((size_t)MR * Dm * 4);
  float* X2   = (float*)wsalloc((size_t)MR * Dm * 4);
  float* X3   = (float*)wsalloc((size_t)MR * Dm * 4);
  float* OUTf = (float*)wsalloc((size_t)MR * Dm * 4);
  float* Qca  = (float*)wsalloc((size_t)MR * Dm * 4);         // also Gb later
  float* KV   = (float*)wsalloc((size_t)Bb * TMEM * 2 * Dm * 4);
  float* offs = (float*)wsalloc((size_t)MR * 2 * 4);
  float* refb = (float*)wsalloc((size_t)MR * 2 * 4);
  int*   ri0  = (int*)wsalloc((size_t)MR * NSs * Rr * 4);
  int*   ri1  = (int*)wsalloc((size_t)MR * NSs * Rr * 4);
  float* rw   = (float*)wsalloc((size_t)MR * NSs * Rr * 4);
  // bf16 mirrors / bf16-only intermediates
  unsigned short* Xb   = (unsigned short*)wsalloc((size_t)MR * Dm * 2);
  unsigned short* AOb  = (unsigned short*)wsalloc((size_t)MR * Dm * 2);
  unsigned short* X2b  = (unsigned short*)wsalloc((size_t)MR * Dm * 2);
  unsigned short* X3b  = (unsigned short*)wsalloc((size_t)MR * Dm * 2);
  unsigned short* OUTb = (unsigned short*)wsalloc((size_t)MR * Dm * 2);
  unsigned short* Hfb  = (unsigned short*)wsalloc((size_t)MR * FFd * 2);
  unsigned short* H1b  = (unsigned short*)wsalloc((size_t)MR * Dm * 2);
  unsigned short* PHb  = (unsigned short*)wsalloc((size_t)MR * Dm * 2);
  unsigned short* memb = (unsigned short*)wsalloc((size_t)Bb * TMEM * Dm * 2);

  float* HS = QKV;                       // score hidden (f32, head-consumed)
  float* H2 = QKV + (size_t)MR * Dm;     // bbox hidden2 (f32, head-consumed)
  float* Gb = Qca;                       // pool MLP layer-1 output (f32)

  // weights -> bf16, memory (feat3) -> bf16 (deterministic, per launch)
  for (int li = 0; li < NLl; ++li)
    for (int s = 0; s < 11; ++s) {
      const int n = kW[s].elems;
      cvt_bf16_kernel<<<(n + 255) / 256, 256, 0, stream>>>(LP(li, kW[s].leaf),
                                                           wbf[li][s], n);
    }
  cvt_bf16_kernel<<<(Bb * TMEM * Dm + 255) / 256, 256, 0, stream>>>(
      f3, memb, Bb * TMEM * Dm);

  init_tgt_kernel<<<(MR * Dm + 255) / 256, 256, 0, stream>>>(qe, X, Xb);

  float* prop_base  = (float*)d_out;                        // [NL,B,NQ,2]
  float* score_base = (float*)d_out + (size_t)NLl * MR * 2; // [NL,B,NQ]
  const float scale = 0.17677669529663689f;                 // 1/sqrt(32)

  auto gemm = [&](const unsigned short* Aa, const unsigned short* Ww,
                  const float* bb, float* Cc, unsigned short* Cbb,
                  int M, int N, int K, bool relu, bool outbf) {
    const int blocks = ((M / 32) * (N / 64) + 7) / 8;
    if (outbf)
      gemm_bf16_kernel<1, 1><<<blocks, 256, 0, stream>>>(Aa, Ww, bb, Cc, Cbb, M, N, K);
    else if (relu)
      gemm_bf16_kernel<1, 0><<<blocks, 256, 0, stream>>>(Aa, Ww, bb, Cc, Cbb, M, N, K);
    else
      gemm_bf16_kernel<0, 0><<<blocks, 256, 0, stream>>>(Aa, Ww, bb, Cc, Cbb, M, N, K);
  };
  auto addln = [&](const float* Xa, const float* Pp, const float* g,
                   const float* b, float* O, unsigned short* Ob) {
    add_ln_kernel<<<MR / 8, 256, 0, stream>>>(Xa, Pp, g, b, O, Ob, MR);
  };

  for (int li = 0; li < NLl; ++li) {
    // ---- self attention ----
    gemm(Xb, wbf[li][0], LP(li, 24), QKV, nullptr, MR, 3 * Dm, Dm, false, false);
    {
      dim3 g(Bb * NHh, (NQq + 127) / 128);
      attn_kernel<<<g, 128, 0, stream>>>(QKV, 3 * Dm, QKV + Dm, 3 * Dm,
                                         QKV + 2 * Dm, 3 * Dm, AOb, Dm,
                                         NQq, NQq, scale);
    }
    gemm(AOb, wbf[li][1], LP(li, 26), P, nullptr, MR, Dm, Dm, false, false);
    addln(X, P, LP(li, 15), LP(li, 14), X2, X2b);

    // ---- cross attention (memory = feat3) ----
    gemm(X2b, wbf[li][2], LP(li, 6), Qca, nullptr, MR, Dm, Dm, false, false);
    gemm(memb, wbf[li][2] + (size_t)Dm * Dm, LP(li, 6) + Dm, KV, nullptr,
         Bb * TMEM, 2 * Dm, Dm, false, false);
    {
      dim3 g(Bb * NHh, (NQq + 127) / 128);
      attn_kernel<<<g, 128, 0, stream>>>(Qca, Dm, KV, 2 * Dm, KV + Dm, 2 * Dm,
                                         AOb, Dm, NQq, TMEM, scale);
    }
    gemm(AOb, wbf[li][3], LP(li, 8), P, nullptr, MR, Dm, Dm, false, false);
    addln(X2, P, LP(li, 17), LP(li, 16), X3, X3b);

    // ---- FFN ----
    gemm(X3b, wbf[li][4], LP(li, 10), nullptr, Hfb, MR, FFd, Dm, true, true);
    gemm(Hfb, wbf[li][5], LP(li, 12), P, nullptr, MR, Dm, FFd, false, false);
    addln(X3, P, LP(li, 19), LP(li, 18), OUTf, OUTb);

    // ---- score head ----
    gemm(OUTb, wbf[li][6], LP(li, 28), HS, nullptr, MR, Dm, Dm, true, false);
    head_kernel<<<MR / 8, 256, 0, stream>>>(HS, LP(li, 31), LP(li, 30),
                                            score_base + (size_t)li * MR, MR, 1);

    // ---- bbox head ----
    gemm(OUTb, wbf[li][7], LP(li, 0), nullptr, H1b, MR, Dm, Dm, true, true);
    gemm(H1b, wbf[li][8], LP(li, 2), H2, nullptr, MR, Dm, Dm, true, false);
    head_kernel<<<MR / 8, 256, 0, stream>>>(H2, LP(li, 5), LP(li, 4), offs, MR, 2);

    // ---- reference / proposals / ROI tables ----
    ref_update_kernel<<<(MR + 255) / 256, 256, 0, stream>>>(
        offs, refc, refb, prop_base + (size_t)li * MR * 2, ri0, ri1, rw,
        li == 0 ? 1 : 0);

    // ---- fused ROI-align + pool GEMM (dominant op, L2-resident features) ----
    pool_gemm_kernel<<<((MR / 16) * (Dm / 64) + 7) / 8, 256, 0, stream>>>(
        f0, f1, f2, f3, ri0, ri1, rw, wbf[li][9], LP(li, 20), PHb);
    gemm(PHb, wbf[li][10], LP(li, 22), Gb, nullptr, MR, Dm, Dm, false, false);
    add_qe_kernel<<<(MR * Dm + 255) / 256, 256, 0, stream>>>(Gb, qe, X, Xb);
  }
}
